// EdgeMLP_5652176962164
// MI455X (gfx1250) — matrix operand order repaired and checked by reading, not compiled, not run on hardware
//
#include <hip/hip_runtime.h>

#define Bc 4
#define Nn 384
#define Dd 128
#define TI 8
#define TJ 16

typedef __attribute__((ext_vector_type(8)))  float  v8f;
typedef __attribute__((ext_vector_type(16))) __bf16 v16bf;

union Frag { uint4 q[2]; v16bf v; };

__device__ __forceinline__ unsigned short f2bf(float f) {
  unsigned int u = __float_as_uint(f);
  u += 0x7fffu + ((u >> 16) & 1u);   // round-to-nearest-even
  return (unsigned short)(u >> 16);
}

// Branch-free erf-based GELU (A&S 7.1.26, |err| <= 1.5e-7): avoids libm erff's
// exec-divergent range-split paths; uses v_rcp_f32 + v_exp_f32.
__device__ __forceinline__ float gelu_exact(float x) {
  float xs = x * 0.70710678118f;   // x / sqrt(2)
  float ax = fabsf(xs);
  float t  = __builtin_amdgcn_rcpf(fmaf(0.3275911f, ax, 1.0f));
  float poly = t * fmaf(t, fmaf(t, fmaf(t, fmaf(t, 1.061405429f, -1.453152027f),
                                        1.421413741f), -0.284496736f), 0.254829592f);
  float er = fmaf(-poly, __expf(-ax * ax), 1.0f);
  er = copysignf(er, xs);
  return 0.5f * x * (1.0f + er);
}

// Async global -> LDS 16B copy (CDNA5 GLOBAL_LOAD_ASYNC_TO_LDS_B128, ASYNCcnt).
// Per ISA 10.2, the low 32 bits of a generic LDS pointer are the LDS byte address.
__device__ __forceinline__ void async_b128(const void* g, void* l) {
  unsigned loff = (unsigned)(uintptr_t)l;
  asm volatile("global_load_async_to_lds_b128 %0, %1, off"
               :: "v"(loff), "v"(g) : "memory");
}
#define ASYNC_DRAIN() asm volatile("s_wait_asynccnt 0x0" ::: "memory")

// ---------------- kernel 1: row/col means over edge_features ----------------
__global__ __launch_bounds__(128) void means_kernel(const float* __restrict__ edge,
                                                    float* __restrict__ rowmean,
                                                    float* __restrict__ colmean) {
  const int bi = blockIdx.x;               // b*Nn + x
  const int b  = bi / Nn, x = bi - b * Nn;
  const int t  = threadIdx.x;              // d index
  float acc = 0.f;
  if (blockIdx.y == 0) {
    const float* p = edge + ((size_t)bi * Nn) * Dd + t;            // edge[b,x,j,t]
    for (int j = 0; j < Nn; ++j) acc += p[(size_t)j * Dd];
    rowmean[(size_t)bi * Dd + t] = acc * (1.f / Nn);
  } else {
    const float* p = edge + ((size_t)b * Nn * Nn + x) * Dd + t;    // edge[b,i,x,t]
    for (int i = 0; i < Nn; ++i) acc += p[(size_t)i * Nn * Dd];
    colmean[(size_t)bi * Dd + t] = acc * (1.f / Nn);
  }
}

// ---- kernel 2: u[b,i] = node@W_s + b_s + rowmean@W_er ; v[b,j] = node@W_t + b_t + colmean@W_ec ----
__global__ __launch_bounds__(128) void uv_kernel(
    const float* __restrict__ node, const float* __restrict__ rowmean,
    const float* __restrict__ colmean,
    const float* __restrict__ W_s, const float* __restrict__ b_s,
    const float* __restrict__ W_t, const float* __restrict__ b_t,
    const float* __restrict__ W_er, const float* __restrict__ W_ec,
    float* __restrict__ u, float* __restrict__ v) {
  __shared__ float xn[Dd], xm[Dd];
  const int bi = blockIdx.x;
  const int o  = threadIdx.x;
  const int dov = blockIdx.y;
  xn[o] = node[(size_t)bi * Dd + o];
  xm[o] = (dov ? colmean : rowmean)[(size_t)bi * Dd + o];
  __syncthreads();
  const float* Wa = dov ? W_t : W_s;
  const float* Wb = dov ? W_ec : W_er;
  float acc = (dov ? b_t : b_s)[o];
  for (int k = 0; k < Dd; ++k)
    acc = fmaf(xn[k], Wa[k * Dd + o], fmaf(xm[k], Wb[k * Dd + o], acc));
  (dov ? v : u)[(size_t)bi * Dd + o] = acc;
}

// ---- kernel 3: pack W_e (256x128) -> WeT bf16 [n][k], W1 (128x128) -> W1T bf16 [n][k] ----
__global__ __launch_bounds__(256) void pack_kernel(const float* __restrict__ W_e,
                                                   const float* __restrict__ W1,
                                                   unsigned short* __restrict__ WeT,
                                                   unsigned short* __restrict__ W1T) {
  const int idx = blockIdx.x * 256 + threadIdx.x;  // 0..32767
  const int k = idx >> 7, n = idx & 127;
  WeT[n * 256 + k] = f2bf(W_e[idx]);
  if (idx < 128 * 128) W1T[n * 128 + k] = f2bf(W1[idx]);
}

// ---------------- kernel 4: fused edge MLP over an 8i x 16j tile (128 pairs) ----------------
__global__ __launch_bounds__(256) void edge_mlp_kernel(
    const float* __restrict__ edge,
    const float* __restrict__ uvecs,         // s+er per (b,i)
    const float* __restrict__ vvecs,         // t+ec per (b,j)
    const unsigned short* __restrict__ WeT,  // [128][256] bf16 (W_e^T)
    const unsigned short* __restrict__ W1T,  // [128][128] bf16 (W1^T)
    const float* __restrict__ b_e,
    const float* __restrict__ b1,
    float* __restrict__ out) {
  // LDS pool, 55296 B:
  //   [0, 17408)  ushort : lds_H (128 x 128, stride 136)  -- overlays lds_X (stride 40)
  //   [17408, +5120) x2  : double-buffered W tile (128 cols x 32 k, stride 40)
  __shared__ __align__(16) unsigned short lds_pool[27648];
  unsigned short* const lds_X = lds_pool;
  unsigned short* const lds_H = lds_pool;
  unsigned short* const wbuf0 = lds_pool + 17408;
  unsigned short* const wbuf1 = lds_pool + 17408 + 5120;

  const int tid  = threadIdx.x;
  const int wave = tid >> 5;
  const int lane = tid & 31;
  const int h    = lane >> 4;    // lane half (WMMA K split)
  const int l16  = lane & 15;
  const int jt   = blockIdx.x * TJ;
  const int it   = blockIdx.y * TI;
  const int bb   = blockIdx.z;
  const int mt   = wave;         // wave's M tile: pairs m = mt*16+M, i = it+mt, j = jt+M

  const v8f vzero = {0.f, 0.f, 0.f, 0.f, 0.f, 0.f, 0.f, 0.f};
  v8f acc[8];
#pragma unroll
  for (int nt = 0; nt < 8; ++nt) acc[nt] = vzero;

  const int row  = tid >> 1;     // staging row 0..127
  const int half = tid & 1;      // which 16-element half of the 32-wide chunk
  const int ii   = row >> 4, jj = row & 15;
  unsigned short* const wdst = lds_pool + 17408 + row * 40 + half * 16;  // +5120 for buf1

  // prologue: async-stage W_e^T chunk 0 into wbuf0; prefetch transposed edge half
  async_b128(WeT + row * 256 + half * 16, wdst);
  async_b128(WeT + row * 256 + half * 16 + 8, wdst + 8);
  __builtin_prefetch(edge + (((size_t)bb * Nn + (jt + jj)) * Nn + (it + ii)) * Dd + half * 16, 0, 3);

  // ---------------- GEMM1: E = X(128x256) @ W_e(256x128), bf16 WMMA, K chunks of 32 ----------------
  for (int kc = 0; kc < 8; ++kc) {
    __syncthreads();   // all waves done with chunk kc-1 compute (X + both W buffers quiescent)
    {  // stage X chunk synchronously: fp32 global -> bf16 LDS
      const float* src = (kc < 4)
        ? edge + (((size_t)bb * Nn + (it + ii)) * Nn + (jt + jj)) * Dd + kc * 32 + half * 16
        : edge + (((size_t)bb * Nn + (jt + jj)) * Nn + (it + ii)) * Dd + (kc - 4) * 32 + half * 16;
      unsigned int w[8];
#pragma unroll
      for (int q = 0; q < 4; ++q) {
        float4 f = *(const float4*)(src + 4 * q);
        w[2 * q + 0] = (unsigned int)f2bf(f.x) | ((unsigned int)f2bf(f.y) << 16);
        w[2 * q + 1] = (unsigned int)f2bf(f.z) | ((unsigned int)f2bf(f.w) << 16);
      }
      uint4* dst = (uint4*)(lds_X + row * 40 + half * 16);
      dst[0] = make_uint4(w[0], w[1], w[2], w[3]);
      dst[1] = make_uint4(w[4], w[5], w[6], w[7]);
    }
    ASYNC_DRAIN();     // this wave's async W chunk kc has landed in LDS
    __syncthreads();   // X visible; every wave drained -> whole W chunk kc resident
    if (kc + 1 < 8) {  // overlap next W chunk with this chunk's WMMAs
      const unsigned short* g = WeT + row * 256 + (kc + 1) * 32 + half * 16;
      unsigned short* l = wdst + ((kc + 1) & 1) * 5120;
      async_b128(g, l);
      async_b128(g + 8, l + 8);
    }
    const unsigned short* wb = (kc & 1) ? wbuf1 : wbuf0;
    Frag fa;  // A frag: lane%16 = M; VGPR0-3: K=8h..8h+7, VGPR4-7: K=16+8h..+7
    {
      const unsigned short* ap = lds_X + (mt * 16 + l16) * 40;
      fa.q[0] = *(const uint4*)(ap + 8 * h);
      fa.q[1] = *(const uint4*)(ap + 16 + 8 * h);
    }
#pragma unroll
    for (int nt = 0; nt < 8; ++nt) {
      Frag fb;  // B frag: lane%16 = N; VGPR0-3: K=8h..8h+7, VGPR4-7: K=16+8h..+7
      const unsigned short* bp = wb + (nt * 16 + l16) * 40;
      fb.q[0] = *(const uint4*)(bp + 8 * h);
      fb.q[1] = *(const uint4*)(bp + 16 + 8 * h);
      acc[nt] = __builtin_amdgcn_wmma_f32_16x16x32_bf16(false, fa.v, false, fb.v,
                                                        (short)0, acc[nt], false, false);
    }
  }

  __syncthreads();  // GEMM1 fully done everywhere: X region reusable as H, wbuf0 free

  // async-stage W1^T chunk 0 during the epilogue
  async_b128(W1T + row * 128 + half * 16, wdst);
  async_b128(W1T + row * 128 + half * 16 + 8, wdst + 8);

  // ---------------- epilogue 1: + b_e + u[i] + v[j], GELU -> H (bf16, LDS) ----------------
  const int i_g = it + mt;
#pragma unroll
  for (int nt = 0; nt < 8; ++nt) {
    const int col = nt * 16 + l16;
    const float add_i = b_e[col] + uvecs[((size_t)bb * Nn + i_g) * Dd + col];
#pragma unroll
    for (int p = 0; p < 8; ++p) {
      const int M   = p + 8 * h;      // C layout: VGPR p -> M=p (lanes 0-15) / M=p+8 (lanes 16-31)
      const int j_g = jt + M;
      float x = acc[nt][p] + add_i + vvecs[((size_t)bb * Nn + j_g) * Dd + col];
      lds_H[(mt * 16 + M) * 136 + col] = f2bf(gelu_exact(x));
    }
    acc[nt] = vzero;  // reuse accumulators for GEMM2
  }

  // ---------------- GEMM2: OUT = H(128x128) @ W1(128x128) ----------------
  for (int kc = 0; kc < 4; ++kc) {
    ASYNC_DRAIN();     // W1 chunk kc resident (this wave's ops)
    __syncthreads();   // H writes / previous compute done everywhere; W chunk kc published
    if (kc + 1 < 4) {
      const unsigned short* g = W1T + row * 128 + (kc + 1) * 32 + half * 16;
      unsigned short* l = wdst + ((kc + 1) & 1) * 5120;
      async_b128(g, l);
      async_b128(g + 8, l + 8);
    }
    const unsigned short* wb = (kc & 1) ? wbuf1 : wbuf0;
    Frag fa;
    {
      const unsigned short* ap = lds_H + (mt * 16 + l16) * 136 + kc * 32;
      fa.q[0] = *(const uint4*)(ap + 8 * h);
      fa.q[1] = *(const uint4*)(ap + 16 + 8 * h);
    }
#pragma unroll
    for (int nt = 0; nt < 8; ++nt) {
      Frag fb;
      const unsigned short* bp = wb + (nt * 16 + l16) * 40;
      fb.q[0] = *(const uint4*)(bp + 8 * h);
      fb.q[1] = *(const uint4*)(bp + 16 + 8 * h);
      acc[nt] = __builtin_amdgcn_wmma_f32_16x16x32_bf16(false, fa.v, false, fb.v,
                                                        (short)0, acc[nt], false, false);
    }
  }

  // ---------------- epilogue 2: + b1, store fp32 ----------------
#pragma unroll
  for (int nt = 0; nt < 8; ++nt) {
    const int col = nt * 16 + l16;
    const float bias1 = b1[col];
#pragma unroll
    for (int p = 0; p < 8; ++p) {
      const int j_g = jt + p + 8 * h;
      out[(((size_t)bb * Nn + i_g) * Nn + j_g) * Dd + col] = acc[nt][p] + bias1;
    }
  }
}

extern "C" void kernel_launch(void* const* d_in, const int* in_sizes, int n_in,
                              void* d_out, int out_size, void* d_ws, size_t ws_size,
                              hipStream_t stream) {
  const float* node = (const float*)d_in[0];
  const float* edge = (const float*)d_in[1];
  const float* W_e  = (const float*)d_in[2];
  const float* b_e  = (const float*)d_in[3];
  const float* W_s  = (const float*)d_in[4];
  const float* b_s  = (const float*)d_in[5];
  const float* W_t  = (const float*)d_in[6];
  const float* b_t  = (const float*)d_in[7];
  const float* W_er = (const float*)d_in[8];
  const float* W_ec = (const float*)d_in[9];
  const float* W1   = (const float*)d_in[10];
  const float* b1   = (const float*)d_in[11];
  float* out = (float*)d_out;

  char* ws = (char*)d_ws;
  const size_t vec_bytes = (size_t)Bc * Nn * Dd * sizeof(float);  // 786432 B
  float* rowmean = (float*)ws; ws += vec_bytes;
  float* colmean = (float*)ws; ws += vec_bytes;
  float* uvecs   = (float*)ws; ws += vec_bytes;
  float* vvecs   = (float*)ws; ws += vec_bytes;
  unsigned short* WeT = (unsigned short*)ws; ws += (size_t)256 * 128 * sizeof(unsigned short);
  unsigned short* W1T = (unsigned short*)ws; ws += (size_t)128 * 128 * sizeof(unsigned short);

  means_kernel<<<dim3(Bc * Nn, 2), 128, 0, stream>>>(edge, rowmean, colmean);
  uv_kernel<<<dim3(Bc * Nn, 2), 128, 0, stream>>>(node, rowmean, colmean,
                                                  W_s, b_s, W_t, b_t, W_er, W_ec,
                                                  uvecs, vvecs);
  pack_kernel<<<128, 256, 0, stream>>>(W_e, W1, WeT, W1T);
  edge_mlp_kernel<<<dim3(Nn / TJ, Nn / TI, Bc), 256, 0, stream>>>(
      edge, uvecs, vvecs, WeT, W1T, b_e, b1, out);
}
